// MultiHeadAttention_82291573391370
// MI455X (gfx1250) — matrix operand order repaired and checked by reading, not compiled, not run on hardware
//
#include <hip/hip_runtime.h>
#include <hip/hip_bf16.h>
#include <cstdint>

// ---------------------------------------------------------------------------
// Types for CDNA5 WMMA (wave32, 16x16x32 bf16 -> f32 accumulate)
// ---------------------------------------------------------------------------
typedef __bf16 bf16;
typedef __attribute__((ext_vector_type(16))) __bf16 v16bf;
typedef __attribute__((ext_vector_type(8)))  __bf16 v8bf;
typedef __attribute__((ext_vector_type(8)))  float  v8f;

__device__ __forceinline__ v8f wmma_bf16(v16bf a, v16bf b, v8f c) {
  // (neg_a, A, neg_b, B, c_mod, C, reuse_a, reuse_b)
  return __builtin_amdgcn_wmma_f32_16x16x32_bf16(false, a, false, b,
                                                 (short)0, c, false, false);
}

#define BB   4
#define SS   2048
#define DD   1024
#define HH   16
#define DH   64
#define MM   (BB * SS)      // 8192 rows

// ---------------------------------------------------------------------------
// 1) LayerNorm: fp32 x -> bf16 xn  (one block per row)
// ---------------------------------------------------------------------------
__global__ __launch_bounds__(256) void ln_kernel(const float* __restrict__ x,
                                                 const float* __restrict__ gamma,
                                                 const float* __restrict__ beta,
                                                 bf16* __restrict__ xn) {
  const int row = blockIdx.x;
  const float* xr = x + (size_t)row * DD;
  __shared__ float red[2][8];
  float s = 0.f, s2 = 0.f;
  for (int i = threadIdx.x; i < DD; i += 256) { float v = xr[i]; s += v; s2 += v*v; }
  for (int m = 16; m >= 1; m >>= 1) { s += __shfl_xor(s, m); s2 += __shfl_xor(s2, m); }
  const int w = threadIdx.x >> 5, lane = threadIdx.x & 31;
  if (lane == 0) { red[0][w] = s; red[1][w] = s2; }
  __syncthreads();
  if (w == 0) {
    float a = (lane < 8) ? red[0][lane] : 0.f;
    float b = (lane < 8) ? red[1][lane] : 0.f;
    for (int m = 4; m >= 1; m >>= 1) { a += __shfl_xor(a, m); b += __shfl_xor(b, m); }
    if (lane == 0) { red[0][0] = a; red[1][0] = b; }
  }
  __syncthreads();
  const float mu  = red[0][0] * (1.f / DD);
  const float var = red[1][0] * (1.f / DD) - mu * mu;
  const float r   = rsqrtf(var + 1e-5f);
  bf16* xo = xn + (size_t)row * DD;
  for (int i = threadIdx.x; i < DD; i += 256)
    xo[i] = (bf16)((xr[i] - mu) * r * gamma[i] + beta[i]);
}

// ---------------------------------------------------------------------------
// 2) Pack fp32 weight [K,N] row-major into bf16 WMMA B-fragment tiles.
//    Tile = 32(K) x 16(N); within tile, lane L owns 16 contiguous bf16:
//    off = ((k>>4)*16 + (n&15))*16 + (k&15)
// ---------------------------------------------------------------------------
__global__ __launch_bounds__(256) void pack_w(const float* __restrict__ W,
                                              bf16* __restrict__ out,
                                              int K, int N) {
  size_t e = (size_t)blockIdx.x * 256 + threadIdx.x;
  if (e >= (size_t)K * N) return;
  int k = (int)(e / N), n = (int)(e % N);
  int kt = k >> 5, nt = n >> 4, kk = k & 31, nn = n & 15;
  size_t tile = (size_t)kt * (N >> 4) + nt;
  size_t off  = tile * 512 + (size_t)(((kk >> 4) * 16 + nn) * 16 + (kk & 15));
  out[off] = (bf16)W[(size_t)k * N + n];
}

// ---------------------------------------------------------------------------
// 3) WMMA GEMM: C[M,N] = A_bf16[M,K] @ Wpacked + bias, with mode-epilogues.
//    Per-wave tile: 32(M) x 32(N); 4 waves/block.
//    mode 0: write Q as [B,H,S,Dh] bf16
//    mode 1: write K as [B,H,S,Dh] bf16 and V^T as [B,H,Dh,S] bf16
//    mode 2: write fp32 out = acc + bias + residual
// ---------------------------------------------------------------------------
__global__ __launch_bounds__(128) void gemm_ws(const bf16* __restrict__ A,
                                               const bf16* __restrict__ Wp,
                                               const float* __restrict__ bias,
                                               int M, int N, int K, int mode,
                                               bf16* __restrict__ outQ,
                                               bf16* __restrict__ outK,
                                               bf16* __restrict__ outVt,
                                               const float* __restrict__ resid,
                                               float* __restrict__ outF) {
  const int lane = threadIdx.x & 31, wave = threadIdx.x >> 5;
  const int half = lane >> 4, l16 = lane & 15;
  const int tiles_n = N >> 5;
  const int tiles_total = (M >> 5) * tiles_n;
  const int tile = blockIdx.x * 4 + wave;
  if (tile >= tiles_total) return;            // wave-uniform: EXEC stays full
  const int m0 = (tile / tiles_n) * 32;
  const int n0 = (tile % tiles_n) * 32;

  v8f acc[2][2] = {};
  const size_t wp_stride = (size_t)(N >> 4) * 512;
  const bf16* wp0 = Wp + (size_t)(n0 >> 4) * 512 + (size_t)lane * 16;

  for (int kt = 0; kt < (K >> 5); ++kt) {
    const int k0 = kt * 32;
    v16bf a[2], b[2];
    for (int mf = 0; mf < 2; ++mf) {
      const bf16* ap = A + (size_t)(m0 + mf * 16 + l16) * K + k0 + half * 16;
      union { v16bf v; v8bf h[2]; } u;
      u.h[0] = *(const v8bf*)(ap);
      u.h[1] = *(const v8bf*)(ap + 8);
      a[mf] = u.v;
    }
    for (int nf = 0; nf < 2; ++nf)
      b[nf] = *(const v16bf*)(wp0 + (size_t)kt * wp_stride + (size_t)nf * 512);
    for (int mf = 0; mf < 2; ++mf)
      for (int nf = 0; nf < 2; ++nf)
        acc[mf][nf] = wmma_bf16(a[mf], b[nf], acc[mf][nf]);
  }

  for (int mf = 0; mf < 2; ++mf)
    for (int nf = 0; nf < 2; ++nf)
      for (int v = 0; v < 8; ++v) {
        const int m = m0 + mf * 16 + half * 8 + v;
        const int n = n0 + nf * 16 + l16;
        const float val = acc[mf][nf][v] + bias[n];
        const int b_ = m / SS, s = m % SS;
        if (mode == 0) {
          const int h = n >> 6, d = n & 63;
          outQ[(((size_t)b_ * HH + h) * SS + s) * DH + d] = (bf16)val;
        } else if (mode == 1) {
          if (n < HH * DH) {
            const int h = n >> 6, d = n & 63;
            outK[(((size_t)b_ * HH + h) * SS + s) * DH + d] = (bf16)val;
          } else {
            const int n2 = n - HH * DH, h = n2 >> 6, d = n2 & 63;
            outVt[(((size_t)b_ * HH + h) * DH + d) * SS + s] = (bf16)val;
          }
        } else {
          outF[(size_t)m * N + n] = val + resid[(size_t)m * N + n];
        }
      }
}

// ---------------------------------------------------------------------------
// 4) Flash attention. One wave owns a 32(q) x 64(Dh) output tile; streams
//    32-key blocks with online softmax. Per block: 8 WMMA (Q.K^T) + 8 WMMA (P.V).
// ---------------------------------------------------------------------------
__global__ __launch_bounds__(128) void attn_kernel(const bf16* __restrict__ Qb,
                                                   const bf16* __restrict__ Kb,
                                                   const bf16* __restrict__ Vt,
                                                   const int* __restrict__ lengths,
                                                   bf16* __restrict__ attn_out) {
  __shared__ bf16 pbuf[4][32 * 32];
  const int lane = threadIdx.x & 31, wave = threadIdx.x >> 5;
  const int half = lane >> 4, l16 = lane & 15;
  const int qtiles = SS >> 5;
  const int gw = blockIdx.x * 4 + wave;
  if (gw >= BB * HH * qtiles) return;          // wave-uniform
  const int qt = gw % qtiles, bh = gw / qtiles;
  const int b = bh / HH, h = bh % HH;
  const int q0 = qt * 32;
  const int len = lengths[b];
  const bf16* Qh = Qb + ((size_t)b * HH + h) * SS * DH;
  const bf16* Kh = Kb + ((size_t)b * HH + h) * SS * DH;
  const bf16* Vh = Vt + ((size_t)b * HH + h) * DH * SS;
  bf16* lds = pbuf[wave];

  // Q A-frags: [mf][ks], K-dim = Dh = 2 steps of 32
  v16bf aq[2][2];
  for (int mf = 0; mf < 2; ++mf) {
    const int row = q0 + mf * 16 + l16;
    for (int ks = 0; ks < 2; ++ks) {
      const bf16* p = Qh + (size_t)row * DH + ks * 32 + half * 16;
      union { v16bf v; v8bf hh[2]; } u;
      u.hh[0] = *(const v8bf*)p;
      u.hh[1] = *(const v8bf*)(p + 8);
      aq[mf][ks] = u.v;
    }
  }

  v8f acc[2][4] = {};
  float mrow[2][8], lrow[2][8], fac[2][8];
  for (int mf = 0; mf < 2; ++mf)
    for (int v = 0; v < 8; ++v) { mrow[mf][v] = -1e30f; lrow[mf][v] = 0.f; }

  const float scale = 0.125f;                  // 1/sqrt(64)
  for (int kbi = 0; kbi <= qt; ++kbi) {
    const int kb = kbi * 32;
    v8f sf[2][2];                              // [nf][mf]
    for (int nf = 0; nf < 2; ++nf) {
      v16bf bk[2];
      for (int ks = 0; ks < 2; ++ks)
        bk[ks] = *(const v16bf*)(Kh + (size_t)(kb + nf * 16 + l16) * DH +
                                 ks * 32 + half * 16);
      for (int mf = 0; mf < 2; ++mf) {
        v8f c = {};
        c = wmma_bf16(aq[mf][0], bk[0], c);
        c = wmma_bf16(aq[mf][1], bk[1], c);
        sf[nf][mf] = c;
      }
    }
    // scale + causal/pad mask
    for (int nf = 0; nf < 2; ++nf)
      for (int mf = 0; mf < 2; ++mf)
        for (int v = 0; v < 8; ++v) {
          const int rq = q0 + mf * 16 + half * 8 + v;
          const int ck = kb + nf * 16 + l16;
          const float val = sf[nf][mf][v] * scale;
          sf[nf][mf][v] = ((ck <= rq) && (ck < len)) ? val : -1e30f;
        }
    // online softmax stats (row lives across 16 lanes of one half)
    for (int mf = 0; mf < 2; ++mf)
      for (int v = 0; v < 8; ++v) {
        float t = fmaxf(sf[0][mf][v], sf[1][mf][v]);
        for (int msk = 1; msk < 16; msk <<= 1) t = fmaxf(t, __shfl_xor(t, msk));
        const float mn = fmaxf(mrow[mf][v], t);
        fac[mf][v] = __expf(mrow[mf][v] - mn);
        mrow[mf][v] = mn;
      }
    for (int mf = 0; mf < 2; ++mf)
      for (int v = 0; v < 8; ++v) {
        const float p0 = __expf(sf[0][mf][v] - mrow[mf][v]);
        const float p1 = __expf(sf[1][mf][v] - mrow[mf][v]);
        sf[0][mf][v] = p0; sf[1][mf][v] = p1;
        float rs = p0 + p1;
        for (int msk = 1; msk < 16; msk <<= 1) rs += __shfl_xor(rs, msk);
        lrow[mf][v] = lrow[mf][v] * fac[mf][v] + rs;
      }
    // P: C-layout -> LDS (row-major 32x32) -> A-layout
    for (int nf = 0; nf < 2; ++nf)
      for (int mf = 0; mf < 2; ++mf)
        for (int v = 0; v < 8; ++v)
          lds[(mf * 16 + half * 8 + v) * 32 + nf * 16 + l16] = (bf16)sf[nf][mf][v];
    asm volatile("s_wait_dscnt 0" ::: "memory");
    for (int mf = 0; mf < 2; ++mf)
      for (int nt = 0; nt < 4; ++nt)
        for (int v = 0; v < 8; ++v) acc[mf][nt][v] *= fac[mf][v];
    v16bf ap[2];
    for (int mf = 0; mf < 2; ++mf) {
      const int row = mf * 16 + l16;
      union { v16bf v; v8bf hh[2]; } u;
      u.hh[0] = *(const v8bf*)(lds + row * 32 + half * 16);
      u.hh[1] = *(const v8bf*)(lds + row * 32 + 8 + half * 16);
      ap[mf] = u.v;
    }
    for (int nt = 0; nt < 4; ++nt) {
      v16bf bv = *(const v16bf*)(Vh + (size_t)(nt * 16 + l16) * SS + kb + half * 16);
      for (int mf = 0; mf < 2; ++mf)
        acc[mf][nt] = wmma_bf16(ap[mf], bv, acc[mf][nt]);
    }
  }
  // normalize + write [B*S, H*Dh] bf16 for the O-projection GEMM
  for (int mf = 0; mf < 2; ++mf)
    for (int nt = 0; nt < 4; ++nt)
      for (int v = 0; v < 8; ++v) {
        const int rq = q0 + mf * 16 + half * 8 + v;
        const float val = acc[mf][nt][v] / lrow[mf][v];
        attn_out[((size_t)b * SS + rq) * (HH * DH) + h * DH + nt * 16 + l16] = (bf16)val;
      }
}

// ---------------------------------------------------------------------------
// Launch
// ---------------------------------------------------------------------------
extern "C" void kernel_launch(void* const* d_in, const int* in_sizes, int n_in,
                              void* d_out, int out_size, void* d_ws, size_t ws_size,
                              hipStream_t stream) {
  const float* x     = (const float*)d_in[0];
  const int*   lens  = (const int*)  d_in[1];
  const float* Wq    = (const float*)d_in[2];
  const float* bq    = (const float*)d_in[3];
  const float* Wkv   = (const float*)d_in[4];
  const float* bkv   = (const float*)d_in[5];
  const float* Wo    = (const float*)d_in[6];
  const float* bo    = (const float*)d_in[7];
  const float* gamma = (const float*)d_in[8];
  const float* beta  = (const float*)d_in[9];
  float* out = (float*)d_out;

  char* ws = (char*)d_ws;
  const size_t MB = 1024 * 1024;
  bf16* xn    = (bf16*)(ws + 0 * MB);    // 16 MiB  [M, D]
  bf16* Wq_p  = (bf16*)(ws + 16 * MB);   //  2 MiB
  bf16* Wkv_p = (bf16*)(ws + 18 * MB);   //  4 MiB
  bf16* Wo_p  = (bf16*)(ws + 22 * MB);   //  2 MiB
  bf16* Qb    = (bf16*)(ws + 24 * MB);   // 16 MiB  [B,H,S,Dh]
  bf16* Kb    = (bf16*)(ws + 40 * MB);   // 16 MiB  [B,H,S,Dh]
  bf16* Vt    = (bf16*)(ws + 56 * MB);   // 16 MiB  [B,H,Dh,S]
  bf16* attn  = (bf16*)(ws + 72 * MB);   // 16 MiB  [M, H*Dh]

  ln_kernel<<<MM, 256, 0, stream>>>(x, gamma, beta, xn);

  pack_w<<<(DD * DD + 255) / 256, 256, 0, stream>>>(Wq,  Wq_p,  DD, DD);
  pack_w<<<(DD * 2 * DD + 255) / 256, 256, 0, stream>>>(Wkv, Wkv_p, DD, 2 * DD);
  pack_w<<<(DD * DD + 255) / 256, 256, 0, stream>>>(Wo,  Wo_p,  DD, DD);

  // Q projection: 8192x1024x1024
  gemm_ws<<<(MM / 32) * (DD / 32) / 4, 128, 0, stream>>>(
      xn, Wq_p, bq, MM, DD, DD, 0, Qb, nullptr, nullptr, nullptr, nullptr);
  // KV projection: 8192x2048x1024 (split epilogue: K row-major, V transposed)
  gemm_ws<<<(MM / 32) * (2 * DD / 32) / 4, 128, 0, stream>>>(
      xn, Wkv_p, bkv, MM, 2 * DD, DD, 1, nullptr, Kb, Vt, nullptr, nullptr);

  // Flash attention: B*H*(S/32) = 4096 waves
  attn_kernel<<<BB * HH * (SS / 32) / 4, 128, 0, stream>>>(Qb, Kb, Vt, lens, attn);

  // Output projection + bias + residual -> fp32 d_out
  gemm_ws<<<(MM / 32) * (DD / 32) / 4, 128, 0, stream>>>(
      attn, Wo_p, bo, MM, DD, DD, 2, nullptr, nullptr, nullptr, x, out);
}
